// Graph_MERFISH_31542239822514
// MI455X (gfx1250) — matrix operand order repaired and checked by reading, not compiled, not run on hardware
//
#include <hip/hip_runtime.h>
#include <hip/hip_bf16.h>

// ---------------- problem constants (match reference) ----------------
#define NN   50000
#define EE   800000
#define ET   (EE + NN)       // edges + self loops
#define FIN  500
#define KP1  512             // FIN padded to multiple of 32
#define NH_  5               // heads
#define CC   32              // channels per head
#define HID_ 160             // NH_*CC   (already multiple of 32)
#define NCL  20
#define GG   8
#define SLOPE_ATT 0.2f
#define SLOPE_ACT 0.01f

#define BPAD 8               // LDS column padding (bf16 elems = 16B) to spread banks
#define WPB  8               // waves per block in GEMM (256 threads)

typedef __attribute__((ext_vector_type(16))) __bf16 v16bf;
typedef __attribute__((ext_vector_type(8)))  __bf16 v8bf;
typedef __attribute__((ext_vector_type(8)))  float  v8f;

// ---------------- helpers ----------------
__device__ __forceinline__ unsigned ford(float f) {
  unsigned u = __float_as_uint(f);
  return (u & 0x80000000u) ? ~u : (u | 0x80000000u);
}
__device__ __forceinline__ float uord(unsigned u) {
  return (u & 0x80000000u) ? __uint_as_float(u & 0x7fffffffu) : __uint_as_float(~u);
}
__device__ __forceinline__ void edge_sd(const long long* __restrict__ ei, int e, int& s, int& d) {
  if (e < EE) { s = (int)ei[e]; d = (int)ei[EE + e]; }
  else        { s = e - EE; d = e - EE; }           // self loop
}
__device__ __forceinline__ v16bf cat16(v8bf lo, v8bf hi) {
  return __builtin_shufflevector(lo, hi, 0, 1, 2, 3, 4, 5, 6, 7, 8, 9, 10, 11, 12, 13, 14, 15);
}

// ---------------- A: fp32 [M,K] -> bf16 [M,Kp] zero-padded ----------------
__global__ void a_pad_bf16_kernel(const float* __restrict__ A, __bf16* __restrict__ Ab,
                                  int M, int K, int Kp) {
  long i = (long)blockIdx.x * blockDim.x + threadIdx.x;
  if (i >= (long)M * Kp) return;
  int r = (int)(i / Kp), k = (int)(i % Kp);
  Ab[i] = (k < K) ? (__bf16)A[(long)r * K + k] : (__bf16)0.0f;
}

// ---------------- W: fp32 [K,Nd] -> bf16 transposed [Nd,Kp] zero-padded ----------------
__global__ void wtrans_kernel(const float* __restrict__ W, __bf16* __restrict__ Wt,
                              int K, int Nd, int Kp) {
  int i = blockIdx.x * blockDim.x + threadIdx.x;
  if (i >= Nd * Kp) return;
  int n = i / Kp, k = i % Kp;
  Wt[i] = (k < K) ? (__bf16)W[k * Nd + n] : (__bf16)0.0f;
}

// ---------------- WMMA GEMM: C[M,Nd] = act(A@B + bias) ----------------
// A: bf16 [M,Kp] row-major (Kp % 32 == 0, branch-free K loop).
// B: bf16 [Nd,Kp] transposed. A block (8 waves) shares one 32-column n-pair:
// the 32 weight columns are staged once into LDS (<=32KB), then every wave
// streams its own 16 A-rows from global (b128) and builds B fragments from
// LDS (ds_load_b128, padded stride). Two v_wmma_f32_16x16x32_bf16 per K-step
// reuse the A fragment; fp32 accumulation.
__global__ void gemm_bf16_wmma(const __bf16* __restrict__ Ab, const __bf16* __restrict__ Bt,
                               const float* __restrict__ bias, float* __restrict__ Cout,
                               int M, int Kp, int Nd, int hasBias, int hasAct, float slope) {
  __shared__ __bf16 Bs[32 * (KP1 + BPAD)];          // sized for the largest Kp
  const int nPairs = Nd >> 5;
  const int np = blockIdx.x % nPairs;
  const int mb = blockIdx.x / nPairs;
  const int tid = threadIdx.x;
  const int lane = tid & 31;
  const int ldsStride = Kp + BPAD;

  // ---- cooperative LDS fill of the 32 B columns (vectorized, all threads) ----
  {
    const int kVec = Kp >> 3;                        // 8 bf16 per copy
    const __bf16* src = Bt + (long)(np * 32) * Kp;
    for (int i = tid; i < 32 * kVec; i += blockDim.x) {
      int col = i / kVec, kv = i % kVec;
      *(v8bf*)(Bs + col * ldsStride + kv * 8) = *(const v8bf*)(src + (long)col * Kp + kv * 8);
    }
  }
  __syncthreads();

  const int mTiles = M >> 4;
  const int mt = mb * WPB + (tid >> 5);              // wave-uniform
  if (mt >= mTiles) return;                          // whole-wave exit after barrier

  const int row  = mt * 16 + (lane & 15);
  const int col  = lane & 15;                        // local column within the pair
  const int hi   = (lane >> 4) & 1;

  const __bf16* Ap  = Ab + (long)row * Kp;
  const __bf16* Bs0 = Bs + col * ldsStride;
  const __bf16* Bs1 = Bs0 + 16 * ldsStride;

  const int dlo = hi ? 8 : 0;      // A frag: lanes 0-15 K{0..7,16..23}, lanes 16-31 K{8..15,24..31}
  const int dhi = hi ? 24 : 16;
  const int dB  = hi ? 16 : 0;     // B frag: same K order as A frag (dlo/dhi)

  v8f acc0 = {}, acc1 = {};
  for (int k0 = 0; k0 < Kp; k0 += 32) {
    v16bf a  = cat16(*(const v8bf*)(Ap + k0 + dlo), *(const v8bf*)(Ap + k0 + dhi));
    v16bf b0 = cat16(*(const v8bf*)(Bs0 + k0 + dlo), *(const v8bf*)(Bs0 + k0 + dhi));
    v16bf b1 = cat16(*(const v8bf*)(Bs1 + k0 + dlo), *(const v8bf*)(Bs1 + k0 + dhi));
    acc0 = __builtin_amdgcn_wmma_f32_16x16x32_bf16(false, a, false, b0, (short)0, acc0, false, false);
    acc1 = __builtin_amdgcn_wmma_f32_16x16x32_bf16(false, a, false, b1, (short)0, acc1, false, false);
  }

  const int col0 = np * 32 + col;
  const float bv0 = hasBias ? bias[col0] : 0.0f;
  const float bv1 = hasBias ? bias[col0 + 16] : 0.0f;
  const int r0 = mt * 16 + (hi ? 8 : 0);
#pragma unroll
  for (int r = 0; r < 8; ++r) {
    float v0 = acc0[r] + bv0;
    float v1 = acc1[r] + bv1;
    if (hasAct) {
      v0 = (v0 > 0.0f) ? v0 : v0 * slope;
      v1 = (v1 > 0.0f) ? v1 : v1 * slope;
    }
    float* cp = Cout + (long)(r0 + r) * Nd + col0;
    cp[0]  = v0;
    cp[16] = v1;
  }
}

// ---------------- attention coefficients per node ----------------
__global__ void compute_al_kernel(const float* __restrict__ h, const float* __restrict__ a_src,
                                  const float* __restrict__ a_dst,
                                  float* __restrict__ als, float* __restrict__ ald) {
  int idx = blockIdx.x * blockDim.x + threadIdx.x;
  if (idx >= NN * NH_) return;
  int node = idx / NH_, hh = idx % NH_;
  const float* hp = h + (long)node * HID_ + hh * CC;
  const float* as = a_src + hh * CC;
  const float* ad = a_dst + hh * CC;
  float s = 0.0f, d = 0.0f;
#pragma unroll 8
  for (int c = 0; c < CC; ++c) { float v = hp[c]; s += v * as[c]; d += v * ad[c]; }
  als[idx] = s; ald[idx] = d;
}

// ---------------- init kernels ----------------
__global__ void init_seg_kernel(unsigned* __restrict__ mseg, float* __restrict__ sseg) {
  int i = blockIdx.x * blockDim.x + threadIdx.x;
  if (i >= NN * NH_) return;
  mseg[i] = ford(-__builtin_inff());
  sseg[i] = 0.0f;
}
__global__ void init_agg_kernel(float* __restrict__ agg, const float* __restrict__ bias) {
  int i = blockIdx.x * blockDim.x + threadIdx.x;
  if (i >= NN * HID_) return;
  agg[i] = bias[i % HID_];
}
__global__ void init_pool_kernel(float* __restrict__ psum, float* __restrict__ cnt) {
  int i = blockIdx.x * blockDim.x + threadIdx.x;
  if (i < GG * HID_) psum[i] = 0.0f;
  if (i < GG) cnt[i] = 0.0f;
}

// ---------------- edge phase 1: logits + segment max ----------------
__global__ void edge_logits_max_kernel(const long long* __restrict__ ei,
                                       const float* __restrict__ als, const float* __restrict__ ald,
                                       unsigned* __restrict__ mseg, float* __restrict__ stage) {
  int idx = blockIdx.x * blockDim.x + threadIdx.x;
  if (idx >= ET * NH_) return;
  int e = idx / NH_, hh = idx % NH_;
  int s, d; edge_sd(ei, e, s, d);
  float lg = als[s * NH_ + hh] + ald[d * NH_ + hh];
  lg = (lg > 0.0f) ? lg : lg * SLOPE_ATT;
  stage[idx] = lg;
  atomicMax(&mseg[d * NH_ + hh], ford(lg));
}

// ---------------- edge phase 2: exp + segment sum ----------------
__global__ void edge_expsum_kernel(const long long* __restrict__ ei,
                                   const unsigned* __restrict__ mseg, float* __restrict__ sseg,
                                   float* __restrict__ stage) {
  int idx = blockIdx.x * blockDim.x + threadIdx.x;
  if (idx >= ET * NH_) return;
  int e = idx / NH_, hh = idx % NH_;
  int s, d; edge_sd(ei, e, s, d);
  (void)s;
  float ex = __expf(stage[idx] - uord(mseg[d * NH_ + hh]));
  stage[idx] = ex;
  atomicAdd(&sseg[d * NH_ + hh], ex);
}

// ---------------- edge phase 3: normalize + message aggregation ----------------
__global__ void edge_aggregate_kernel(const long long* __restrict__ ei,
                                      const float* __restrict__ sseg,
                                      const float* __restrict__ h,
                                      float* __restrict__ agg, float* __restrict__ alpha) {
  int idx = blockIdx.x * blockDim.x + threadIdx.x;
  if (idx >= ET * NH_) return;
  int e = idx / NH_, hh = idx % NH_;
  int s, d; edge_sd(ei, e, s, d);
  float al = alpha[idx] / (sseg[d * NH_ + hh] + 1e-16f);
  alpha[idx] = al;                                       // final alpha output
  const float* hs = h + (long)s * HID_ + hh * CC;
  float* ag = agg + (long)d * HID_ + hh * CC;
#pragma unroll 8
  for (int c = 0; c < CC; ++c) atomicAdd(&ag[c], hs[c] * al);
}

// ---------------- leaky_relu + layernorm -> bf16, one wave32 per node ----------------
// Output feeds only the next WMMA GEMM, so emit bf16 directly.
__global__ void act_ln_kernel(const float* __restrict__ in, const float* __restrict__ sc,
                              const float* __restrict__ bi, __bf16* __restrict__ outb,
                              float slope) {
  int wave = (blockIdx.x * blockDim.x + threadIdx.x) >> 5;
  int lane = threadIdx.x & 31;
  if (wave >= NN) return;
  const float* p = in + (long)wave * HID_;
  float v[5]; float s = 0.0f, q = 0.0f;
#pragma unroll
  for (int j = 0; j < 5; ++j) {
    float x = p[lane + 32 * j];
    x = (x > 0.0f) ? x : x * slope;
    v[j] = x; s += x; q += x * x;
  }
#pragma unroll
  for (int o = 16; o > 0; o >>= 1) { s += __shfl_xor(s, o, 32); q += __shfl_xor(q, o, 32); }
  float mu = s * (1.0f / HID_);
  float var = q * (1.0f / HID_) - mu * mu;
  float r = rsqrtf(var + 1e-5f);
  __bf16* op = outb + (long)wave * HID_;
#pragma unroll
  for (int j = 0; j < 5; ++j) {
    int ch = lane + 32 * j;
    op[ch] = (__bf16)((v[j] - mu) * r * sc[ch] + bi[ch]);
  }
}

// ---------------- global mean pool: accumulate ----------------
__global__ void pool_accum_kernel(const float* __restrict__ xo, const long long* __restrict__ batch,
                                  float* __restrict__ psum, float* __restrict__ cnt) {
  long idx = (long)blockIdx.x * blockDim.x + threadIdx.x;
  if (idx >= (long)NN * HID_) return;
  int n = (int)(idx / HID_), f = (int)(idx % HID_);
  int b = (int)batch[n];
  atomicAdd(&psum[b * HID_ + f], xo[idx]);
  if (f == 0) atomicAdd(&cnt[b], 1.0f);
}

// ---------------- MLP head (single block): mean, fc1, LN+relu, fc2 ----------------
__global__ void mlp_head_kernel(const float* __restrict__ psum, const float* __restrict__ cnt,
                                const float* __restrict__ W1, const float* __restrict__ b1,
                                const float* __restrict__ lns, const float* __restrict__ lnb,
                                const float* __restrict__ W2, const float* __restrict__ b2,
                                float* __restrict__ rec) {
  __shared__ float pl[GG * HID_];
  __shared__ float zz[GG * HID_];
  __shared__ float mu[GG], rs[GG];
  int tid = threadIdx.x;
  for (int i = tid; i < GG * HID_; i += blockDim.x) {
    int g = i / HID_;
    pl[i] = psum[i] / fmaxf(cnt[g], 1.0f);
  }
  __syncthreads();
  for (int i = tid; i < GG * HID_; i += blockDim.x) {
    int g = i / HID_, f = i % HID_;
    float a = b1[f];
    for (int k = 0; k < HID_; ++k) a += pl[g * HID_ + k] * W1[k * HID_ + f];
    zz[i] = a;
  }
  __syncthreads();
  if (tid < GG) {
    float s = 0.0f, q = 0.0f;
    for (int f = 0; f < HID_; ++f) { float v = zz[tid * HID_ + f]; s += v; q += v * v; }
    float m = s * (1.0f / HID_);
    mu[tid] = m; rs[tid] = rsqrtf(q * (1.0f / HID_) - m * m + 1e-5f);
  }
  __syncthreads();
  for (int i = tid; i < GG * HID_; i += blockDim.x) {
    int g = i / HID_, f = i % HID_;
    float v = (zz[i] - mu[g]) * rs[g] * lns[f] + lnb[f];
    zz[i] = fmaxf(v, 0.0f);
  }
  __syncthreads();
  for (int i = tid; i < GG * NCL; i += blockDim.x) {
    int g = i / NCL, c = i % NCL;
    float a = b2[c];
    for (int k = 0; k < HID_; ++k) a += zz[g * HID_ + k] * W2[k * NCL + c];
    rec[i] = a;
  }
}

// ---------------- launcher ----------------
extern "C" void kernel_launch(void* const* d_in, const int* in_sizes, int n_in,
                              void* d_out, int out_size, void* d_ws, size_t ws_size,
                              hipStream_t stream) {
  (void)in_sizes; (void)n_in; (void)out_size; (void)ws_size;
  const float*     x      = (const float*)d_in[0];
  const long long* ei     = (const long long*)d_in[1];
  const long long* batch  = (const long long*)d_in[2];
  const float* W1     = (const float*)d_in[3];
  const float* a_src1 = (const float*)d_in[4];
  const float* a_dst1 = (const float*)d_in[5];
  const float* b1     = (const float*)d_in[6];
  const float* ln1s   = (const float*)d_in[7];
  const float* ln1b   = (const float*)d_in[8];
  const float* W2     = (const float*)d_in[9];
  const float* a_src2 = (const float*)d_in[10];
  const float* a_dst2 = (const float*)d_in[11];
  const float* b2     = (const float*)d_in[12];
  const float* ln2s   = (const float*)d_in[13];
  const float* ln2b   = (const float*)d_in[14];
  const float* Wm     = (const float*)d_in[15];
  const float* bm     = (const float*)d_in[16];
  const float* Wmlp1  = (const float*)d_in[17];
  const float* bmlp1  = (const float*)d_in[18];
  const float* ln3s   = (const float*)d_in[19];
  const float* ln3b   = (const float*)d_in[20];
  const float* Wmlp2  = (const float*)d_in[21];
  const float* bmlp2  = (const float*)d_in[22];

  // outputs: xo[N,HID], rec[G,NC], alpha1[Et,H], alpha2[Et,H]
  float* out    = (float*)d_out;
  float* xo     = out;
  float* rec    = out + (long)NN * HID_;
  float* alpha1 = rec + GG * NCL;
  float* alpha2 = alpha1 + (long)ET * NH_;

  // workspace layout (~120 MB). Region0 is Ab for GEMM1, then reused as houtb.
  float* ws      = (float*)d_ws;
  __bf16* Ab     = (__bf16*)ws;                       // N*KP1 bf16 = 51.2 MB
  __bf16* houtb  = (__bf16*)ws;                       // N*HID bf16 (reuse after GEMM1)
  float* hfeat   = ws + ((long)NN * KP1) / 2;         // N*HID fp32
  float* agg     = hfeat + (long)NN * HID_;           // N*HID fp32
  float* als     = agg + (long)NN * HID_;             // N*H
  float* ald     = als + NN * NH_;                    // N*H
  unsigned* mseg = (unsigned*)(ald + NN * NH_);       // N*H
  float* sseg    = (float*)(mseg + NN * NH_);         // N*H
  float* psum    = sseg + NN * NH_;                   // G*HID
  float* cnt     = psum + GG * HID_;                  // G
  __bf16* Wt     = (__bf16*)(cnt + GG);               // Nd*Kp bf16 (reused per weight)

  const int BS = 256;
  auto cdiv = [](long a, long b) { return (int)((a + b - 1) / b); };

  const int mTiles   = NN / 16;                                   // 3125
  const int nPairs   = HID_ / 32;                                 // 5
  const int gemmGrid = nPairs * cdiv(mTiles, WPB);                // blocks (8 waves each)
  const int edgeGrid = cdiv((long)ET * NH_, BS);
  const int nodeH    = cdiv((long)NN * NH_, BS);
  const int nodeHid  = cdiv((long)NN * HID_, BS);
  const int lnGrid   = cdiv((long)NN * 32, BS);

  // ===== layer 1 =====
  a_pad_bf16_kernel<<<cdiv((long)NN * KP1, BS), BS, 0, stream>>>(x, Ab, NN, FIN, KP1);
  wtrans_kernel<<<cdiv((long)HID_ * KP1, BS), BS, 0, stream>>>(W1, Wt, FIN, HID_, KP1);
  gemm_bf16_wmma<<<gemmGrid, BS, 0, stream>>>(Ab, Wt, nullptr, hfeat, NN, KP1, HID_, 0, 0, 0.0f);
  compute_al_kernel<<<nodeH, BS, 0, stream>>>(hfeat, a_src1, a_dst1, als, ald);
  init_seg_kernel<<<nodeH, BS, 0, stream>>>(mseg, sseg);
  init_agg_kernel<<<nodeHid, BS, 0, stream>>>(agg, b1);
  edge_logits_max_kernel<<<edgeGrid, BS, 0, stream>>>(ei, als, ald, mseg, alpha1);
  edge_expsum_kernel<<<edgeGrid, BS, 0, stream>>>(ei, mseg, sseg, alpha1);
  edge_aggregate_kernel<<<edgeGrid, BS, 0, stream>>>(ei, sseg, hfeat, agg, alpha1);
  act_ln_kernel<<<lnGrid, BS, 0, stream>>>(agg, ln1s, ln1b, houtb, SLOPE_ACT);

  // ===== layer 2 =====
  wtrans_kernel<<<cdiv((long)HID_ * HID_, BS), BS, 0, stream>>>(W2, Wt, HID_, HID_, HID_);
  gemm_bf16_wmma<<<gemmGrid, BS, 0, stream>>>(houtb, Wt, nullptr, hfeat, NN, HID_, HID_, 0, 0, 0.0f);
  compute_al_kernel<<<nodeH, BS, 0, stream>>>(hfeat, a_src2, a_dst2, als, ald);
  init_seg_kernel<<<nodeH, BS, 0, stream>>>(mseg, sseg);
  init_agg_kernel<<<nodeHid, BS, 0, stream>>>(agg, b2);
  edge_logits_max_kernel<<<edgeGrid, BS, 0, stream>>>(ei, als, ald, mseg, alpha2);
  edge_expsum_kernel<<<edgeGrid, BS, 0, stream>>>(ei, mseg, sseg, alpha2);
  edge_aggregate_kernel<<<edgeGrid, BS, 0, stream>>>(ei, sseg, hfeat, agg, alpha2);
  act_ln_kernel<<<lnGrid, BS, 0, stream>>>(agg, ln2s, ln2b, houtb, SLOPE_ACT);

  // ===== xo = leaky_relu(h2 @ Wm + bm) ===== (fused epilogue)
  wtrans_kernel<<<cdiv((long)HID_ * HID_, BS), BS, 0, stream>>>(Wm, Wt, HID_, HID_, HID_);
  gemm_bf16_wmma<<<gemmGrid, BS, 0, stream>>>(houtb, Wt, bm, xo, NN, HID_, HID_, 1, 1, SLOPE_ACT);

  // ===== global mean pool + MLP head =====
  init_pool_kernel<<<cdiv(GG * HID_ + GG, BS), BS, 0, stream>>>(psum, cnt);
  pool_accum_kernel<<<nodeHid, BS, 0, stream>>>(xo, batch, psum, cnt);
  mlp_head_kernel<<<1, BS, 0, stream>>>(psum, cnt, Wmlp1, bmlp1, ln3s, ln3b, Wmlp2, bmlp2, rec);
}
